// GroupConv_30562987278752
// MI455X (gfx1250) — matrix operand order repaired and checked by reading, not compiled, not run on hardware
//
#include <hip/hip_runtime.h>

// GroupConv as implicit GEMM on CDNA5 WMMA (bf16x3 split-precision, f32 accumulate).
// x:[16,112,112,256] f32, w:[8,3,3,32,32] f32, bias:[256] f32 -> out:[16,112,112,256] f32.
//
// One workgroup per (b, h-quad, g): 6 input rows + the group's 9x32x32 weights are
// staged ONCE into LDS, pre-split into bf16 hi/lo and zero-padded. Each wave computes
// a 16(M)x32(N) tile for 4 consecutive output rows; B fragments are loaded once per
// tap and reused across the 4 rows -> 54 ds_load_b128 per output row vs 54 WMMAs.

typedef __bf16 v16bf __attribute__((ext_vector_type(16)));
typedef __bf16 v8bf  __attribute__((ext_vector_type(8)));
typedef __bf16 v4bf  __attribute__((ext_vector_type(4)));
typedef float  v8f   __attribute__((ext_vector_type(8)));

#define BB   16
#define HH   112
#define WW   112
#define CC   256
#define GG   8
#define PP   32      // in-channels per group
#define FF   32      // out-channels per group
#define TAPS 9
#define NTHREADS 224 // 7 waves of 32; W = 7 * 16 M-tiles
#define HQ   4       // output rows per workgroup (112 = 28 * 4)
#define XROWS (HQ + 2)

#define XW   114     // staged width: w = -1 .. 112 (zero-padded edges)
#define XSTR 40      // p-stride (bf16): 80B = 20 banks -> conflict-free b128
#define WSTR 40

__device__ __forceinline__ void bf16_split(float f, __bf16& hi, __bf16& lo) {
    hi = (__bf16)f;                 // RNE to bf16 (8 mantissa bits)
    lo = (__bf16)(f - (float)hi);   // next 8 mantissa bits
}

__global__ __launch_bounds__(NTHREADS)
void GroupConv_wmma_bf16x3(const float* __restrict__ x,
                           const float* __restrict__ w,
                           const float* __restrict__ bias,
                           float* __restrict__ out)
{
    // [hi/lo][row 0..5][wc 0..113][p 0..31 (pad to 40)]
    __shared__ __align__(16) __bf16 sX[2 * XROWS * XW * XSTR];
    // [hi/lo][tap][f][p (pad to 40)]
    __shared__ __align__(16) __bf16 sW[2 * TAPS * FF * WSTR];

    const int tid  = threadIdx.x;
    const int bid  = blockIdx.x;          // grid = B * (H/HQ) * G
    const int g    = bid % GG;
    const int rest = bid / GG;
    const int h0   = (rest % (HH / HQ)) * HQ;
    const int b    = rest / (HH / HQ);

    // ---- stage + split x tile: rows h0-1 .. h0+HQ, w=-1..112, 32 group channels ----
    for (int i = tid; i < XROWS * XW * (PP / 4); i += NTHREADS) {
        const int p4 = i & 7;
        const int wc = (i >> 3) % XW;
        const int r  = i / (XW * 8);
        const int ih = h0 + r - 1;
        const int iw = wc - 1;
        float4 v;
        if ((unsigned)ih < (unsigned)HH && (unsigned)iw < (unsigned)WW) {
            v = *(const float4*)(x + (((size_t)(b * HH + ih) * WW + iw) * CC + g * PP + p4 * 4));
        } else {
            v = make_float4(0.f, 0.f, 0.f, 0.f);
        }
        union { v4bf v; __bf16 e[4]; } hi, lo;
        const float vv[4] = {v.x, v.y, v.z, v.w};
        #pragma unroll
        for (int j = 0; j < 4; ++j) {
            __bf16 h16, l16;
            bf16_split(vv[j], h16, l16);
            hi.e[j] = h16;
            lo.e[j] = l16;
        }
        const int base = (r * XW + wc) * XSTR + p4 * 4;
        *(v4bf*)(&sX[base])                     = hi.v;   // ds_store_b64
        *(v4bf*)(&sX[base + XROWS * XW * XSTR]) = lo.v;
    }

    // ---- stage + split + transpose weights: global [tap][p][f] -> LDS [tap][f][p] ----
    for (int i = tid; i < TAPS * PP * (FF / 4); i += NTHREADS) {
        const int f4  = i & 7;
        const int p   = (i >> 3) % PP;
        const int tap = i / (PP * 8);
        float4 v = *(const float4*)(w + ((size_t)(g * TAPS + tap) * PP + p) * FF + f4 * 4);
        const float vv[4] = {v.x, v.y, v.z, v.w};
        #pragma unroll
        for (int j = 0; j < 4; ++j) {
            __bf16 h16, l16;
            bf16_split(vv[j], h16, l16);
            const int f = f4 * 4 + j;
            sW[(tap * FF + f) * WSTR + p]                    = h16;
            sW[(tap * FF + f) * WSTR + p + TAPS * FF * WSTR] = l16;
        }
    }
    __syncthreads();

    const int lane = tid & 31;
    const int wave = tid >> 5;       // 0..6
    const int w0   = wave * 16;      // M tile base (w coordinate)
    const int m    = lane & 15;      // A row within tile / C column
    const int half = lane >> 4;      // lane half selects K runs
    const int n    = lane & 15;

    union AF { v16bf v; v8bf h2[2]; };
    union BF { v16bf v; v8bf h2[2]; };
    union CF { v8f v; float e[8]; };

    // acc[row][ntile] -> 8 independent WMMA chains
    CF acc[HQ][2];
    #pragma unroll
    for (int s = 0; s < HQ; ++s) { acc[s][0].v = {}; acc[s][1].v = {}; }

    // Lane-dependent LDS bases computed once; all loop deltas are DS immediates.
    // A (16-bit A 16x32): lane<16 -> K 0..7 & 16..23 ; lane>=16 -> K 8..15 & 24..31
    const __bf16* aHi = &sX[(size_t)(w0 + m) * XSTR + half * 8];
    const __bf16* aLo = aHi + XROWS * XW * XSTR;   // lo region needs its own base (16-bit DS offset)
    // B (16-bit B 32x16): lane<16 -> K 0..7 & 16..23 ; lane>=16 -> K 8..15 & 24..31 at column f = nt*16 + n
    const __bf16* bHi = &sW[(size_t)n * WSTR + half * 8];

    #pragma unroll
    for (int kh = 0; kh < 3; ++kh) {
        #pragma unroll
        for (int kw = 0; kw < 3; ++kw) {
            const int tap = kh * 3 + kw;

            // B fragments: loaded once per tap, reused across the 4 output rows.
            BF Bhi[2], Blo[2];
            #pragma unroll
            for (int nt = 0; nt < 2; ++nt) {
                const int boff = (tap * FF + nt * 16) * WSTR;
                Bhi[nt].h2[0] = *(const v8bf*)(bHi + boff);
                Bhi[nt].h2[1] = *(const v8bf*)(bHi + boff + 16);
                Blo[nt].h2[0] = *(const v8bf*)(bHi + boff + TAPS * FF * WSTR);
                Blo[nt].h2[1] = *(const v8bf*)(bHi + boff + TAPS * FF * WSTR + 16);
            }

            #pragma unroll
            for (int s = 0; s < HQ; ++s) {
                const int aoff = ((s + kh) * XW + kw) * XSTR;   // staged row s+kh, col shift kw
                AF Ahi, Alo;
                Ahi.h2[0] = *(const v8bf*)(aHi + aoff);          // p = half*8
                Ahi.h2[1] = *(const v8bf*)(aHi + aoff + 16);     // p = 16+half*8
                Alo.h2[0] = *(const v8bf*)(aLo + aoff);
                Alo.h2[1] = *(const v8bf*)(aLo + aoff + 16);

                #pragma unroll
                for (int nt = 0; nt < 2; ++nt) {
                    acc[s][nt].v = __builtin_amdgcn_wmma_f32_16x16x32_bf16(
                        false, Ahi.v, false, Bhi[nt].v, (short)0, acc[s][nt].v, false, false);
                    acc[s][nt].v = __builtin_amdgcn_wmma_f32_16x16x32_bf16(
                        false, Ahi.v, false, Blo[nt].v, (short)0, acc[s][nt].v, false, false);
                    acc[s][nt].v = __builtin_amdgcn_wmma_f32_16x16x32_bf16(
                        false, Alo.v, false, Bhi[nt].v, (short)0, acc[s][nt].v, false, false);
                }
            }
        }
    }

    // ---- epilogue: add bias, store 4 rows ----
    // C layout: lane n = column, VGPR r -> output row M = half*8 + r
    const float b0 = bias[g * FF + n];
    const float b1 = bias[g * FF + 16 + n];
    #pragma unroll
    for (int s = 0; s < HQ; ++s) {
        #pragma unroll
        for (int r = 0; r < 8; ++r) {
            const int mrow = half * 8 + r;
            float* q = out + (((size_t)(b * HH + (h0 + s)) * WW + (w0 + mrow)) * (GG * FF)
                              + g * FF + n);
            q[0]  = acc[s][0].e[r] + b0;
            q[16] = acc[s][1].e[r] + b1;
        }
    }
}

extern "C" void kernel_launch(void* const* d_in, const int* in_sizes, int n_in,
                              void* d_out, int out_size, void* d_ws, size_t ws_size,
                              hipStream_t stream) {
    (void)in_sizes; (void)n_in; (void)out_size; (void)d_ws; (void)ws_size;
    const float* x    = (const float*)d_in[0];
    const float* w    = (const float*)d_in[1];
    const float* bias = (const float*)d_in[2];
    float* out        = (float*)d_out;

    dim3 grid(BB * (HH / HQ) * GG);  // 3584 workgroups: one per (b, h-quad, g)
    dim3 block(NTHREADS);            // 7 waves; each wave: 16x32 tile x 4 rows
    GroupConv_wmma_bf16x3<<<grid, block, 0, stream>>>(x, w, bias, out);
}
